// GATLayer_10247791968972
// MI455X (gfx1250) — matrix operand order repaired and checked by reading, not compiled, not run on hardware
//
#include <hip/hip_runtime.h>
#include <hip/hip_bf16.h>
#include <math.h>

typedef __attribute__((ext_vector_type(16))) _Float16     v16h;
typedef __attribute__((ext_vector_type(8)))  float        v8f;
typedef __attribute__((ext_vector_type(4)))  unsigned int u32x4;
typedef __attribute__((ext_vector_type(8)))  int          i32x8;
typedef __attribute__((ext_vector_type(4)))  int          i32x4;

#define Bdim  4
#define Ndim  2048
#define Fdim  64      // F_IN == F_OUT == 64
#define Hdim  8
#define ALPHA 0.2f

__device__ __forceinline__ float leaky(float x) { return x >= 0.0f ? x : ALPHA * x; }

// ---------------------------------------------------------------------------
// TDM: DMA one 2D tile [64 rows x 32 cols, f16, row-stride Ndim] from global
// WhT into LDS (packed [64][32]). D# groups per CDNA5 ISA ch.8 (08_async_tensor).
// Issued by one wave; completion via TENSORcnt + workgroup barrier.
// (6-arg builtin form on this toolchain: g0, g1, g2, g3, g4, cpol.)
// ---------------------------------------------------------------------------
__device__ __forceinline__ void tdm_load_tile(const _Float16* gsrc, _Float16* lds_dst)
{
    const unsigned long long ga = (unsigned long long)(uintptr_t)gsrc;
    const unsigned int       la = (unsigned int)(uintptr_t)lds_dst; // LDS aperture: [31:0] = LDS offset

    u32x4 g0;
    g0[0] = 1u;                                              // count=1 (valid user D#)
    g0[1] = la;                                              // lds_addr (bytes)
    g0[2] = (unsigned int)ga;                                // global_addr[31:0]
    g0[3] = (unsigned int)((ga >> 32) & 0x1FFFFFFull)        // global_addr[56:32]
            | 0x80000000u;                                   // type=2 ("image")

    i32x8 g1;
    g1[0] = (int)(1u << 16);        // workgroup_mask=0, data_size=1 (2 bytes)
    g1[1] = (int)(2048u << 16);     // tensor_dim0[15:0] << 16 (= Ndim)
    g1[2] = (int)(64u << 16);       // tensor_dim0 hi = 0 ; tensor_dim1[15:0]=64 << 16
    g1[3] = (int)(32u << 16);       // tensor_dim1 hi = 0 ; tile_dim0 = 32
    g1[4] = 64;                     // tile_dim1 = 64 rows ; tile_dim2 = 0
    g1[5] = 2048;                   // tensor_dim0_stride lo (= Ndim elements)
    g1[6] = 0;                      // stride hi / dim1_stride lo
    g1[7] = 0;

    const i32x4 z4 = {};
    const i32x8 z8 = {};
    __builtin_amdgcn_tensor_load_to_lds(g0, g1, z4, z4, z8, 0);
}

// ---------------------------------------------------------------------------
// Kernel 1: Wh = h @ W + Wb  (per (b,head), 2048x64x64 GEMM via WMMA f16)
// Writes WhT[b,h,o,n] (f16, n-contiguous) and si/sj[b,h,n] (f32).
// Block: 256 threads = 8 waves; wave w = head w for a 16-row tile.
// ---------------------------------------------------------------------------
__global__ void __launch_bounds__(256)
gat_k1_feature(const float* __restrict__ h, const float* __restrict__ W,
               const float* __restrict__ Wb, const float* __restrict__ a1,
               const float* __restrict__ a2,
               _Float16* __restrict__ WhT, float* __restrict__ si,
               float* __restrict__ sj)
{
    const int lane = threadIdx.x & 31;
    const int head = threadIdx.x >> 5;
    const int m    = lane & 15;
    const int hi   = lane >> 4;
    const int b    = blockIdx.x / (Ndim / 16);
    const int i0   = (blockIdx.x % (Ndim / 16)) * 16;

    // A operand: h[b, i0+m, :64] -> two 16x32 f16 chunks (contiguous K: k = 16*hi + s)
    const float* hrow = h + ((size_t)b * Ndim + (i0 + m)) * Fdim;
    v16h A[2];
#pragma unroll
    for (int kk = 0; kk < 2; ++kk) {
        const int klo = kk * 32 + hi * 16;
        const int khi = kk * 32 + hi * 16 + 8;
#pragma unroll
        for (int t = 0; t < 8; ++t) {
            A[kk][t]     = (_Float16)hrow[klo + t];
            A[kk][8 + t] = (_Float16)hrow[khi + t];
        }
    }

    const float* Whead = W + (size_t)head * Fdim * Fdim;
    v8f acc[4];
#pragma unroll
    for (int ot = 0; ot < 4; ++ot) {
        v8f c = {};
        const int col = ot * 16 + m;
#pragma unroll
        for (int kk = 0; kk < 2; ++kk) {
            const int kbase = kk * 32 + hi * 16;
            v16h Bv;
#pragma unroll
            for (int t = 0; t < 16; ++t)
                Bv[t] = (_Float16)Whead[(size_t)(kbase + t) * Fdim + col];
            c = __builtin_amdgcn_wmma_f32_16x16x32_f16(false, A[kk], false, Bv,
                                                       (short)0, c, false, false);
        }
        acc[ot] = c;
    }

    const int bh = b * Hdim + head;
    float s1[8], s2[8];
#pragma unroll
    for (int r = 0; r < 8; ++r) { s1[r] = 0.0f; s2[r] = 0.0f; }

#pragma unroll
    for (int ot = 0; ot < 4; ++ot) {
        const int col  = ot * 16 + m;
        const float bi = Wb[head * Fdim + col];
        const float w1 = a1[head * Fdim + col];
        const float w2 = a2[head * Fdim + col];
        _Float16* wcol = WhT + ((size_t)bh * Fdim + col) * Ndim;
#pragma unroll
        for (int r = 0; r < 8; ++r) {
            const float v  = acc[ot][r] + bi;
            const int  row = i0 + r + hi * 8;
            wcol[row] = (_Float16)v;
            s1[r] += v * w1;
            s2[r] += v * w2;
        }
    }
#pragma unroll
    for (int r = 0; r < 8; ++r) {
#pragma unroll
        for (int off = 8; off >= 1; off >>= 1) {
            s1[r] += __shfl_xor(s1[r], off, 32);
            s2[r] += __shfl_xor(s2[r], off, 32);
        }
    }
    if (m == 0) {
#pragma unroll
        for (int r = 0; r < 8; ++r) {
            const int row = i0 + r + hi * 8;
            si[(size_t)bh * Ndim + row] = s1[r];
            sj[(size_t)bh * Ndim + row] = s2[r];
        }
    }
}

// ---------------------------------------------------------------------------
// Kernel 2: sjmax[b,h] = max_j sj[b,h,j]  (leaky_relu monotonic -> exact
// per-row softmax max = lr(si + sjmax + Ab); enables single-pass softmax).
// ---------------------------------------------------------------------------
__global__ void __launch_bounds__(256)
gat_k2_sjmax(const float* __restrict__ sj, float* __restrict__ sjmax)
{
    __shared__ float red[256];
    const int bh = blockIdx.x;
    float mx = -3.4e38f;
    for (int j = threadIdx.x; j < Ndim; j += 256)
        mx = fmaxf(mx, sj[(size_t)bh * Ndim + j]);
    red[threadIdx.x] = mx;
    __syncthreads();
    for (int s = 128; s > 0; s >>= 1) {
        if (threadIdx.x < s) red[threadIdx.x] = fmaxf(red[threadIdx.x], red[threadIdx.x + s]);
        __syncthreads();
    }
    if (threadIdx.x == 0) sjmax[bh] = red[0];
}

// ---------------------------------------------------------------------------
// Kernel 3: fused attention, TDM-staged. Block = 8 waves, one (b,head), 128
// query rows. Double-buffered LDS tile of WhT[., jc:jc+32] DMA'd by TDM
// (wave 0), all waves read B operands from LDS. Per chunk: p = exp(lr(..)-m)
// built in registers as f16 A-operand; 4 WMMAs; single-pass row sums;
// epilogue: sigmoid(elu(acc / l)) -> [B, N, H*F].
// ---------------------------------------------------------------------------
__global__ void __launch_bounds__(256)
gat_k3_attn(const _Float16* __restrict__ WhT, const float* __restrict__ si,
            const float* __restrict__ sj, const float* __restrict__ sjmax,
            const float* __restrict__ Ab, float* __restrict__ out)
{
    __shared__ __align__(16) _Float16 lbuf[2][Fdim * 32];   // 2 x 4 KB

    const int lane = threadIdx.x & 31;
    const int wave = threadIdx.x >> 5;
    const int m    = lane & 15;
    const int hi   = lane >> 4;

    const int nb   = Ndim / 128;
    const int bh   = blockIdx.x / nb;
    const int rb   = blockIdx.x % nb;
    const int b    = bh / Hdim;
    const int head = bh % Hdim;
    const int i0   = rb * 128 + wave * 16;

    const float ab   = Ab[head];
    const float sim  = si[(size_t)bh * Ndim + i0 + m];
    const float mrow = leaky(sim + sjmax[bh] + ab);

    const float*    sjp    = sj + (size_t)bh * Ndim;
    const _Float16* whbase = WhT + (size_t)bh * Fdim * Ndim;

    v8f acc[4];
#pragma unroll
    for (int ot = 0; ot < 4; ++ot) acc[ot] = (v8f){};
    float lsum = 0.0f;

    // Prologue: DMA first tile
    if (wave == 0) tdm_load_tile(whbase, lbuf[0]);

    int cur = 0;
    for (int jc = 0; jc < Ndim; jc += 32) {
        const bool havenext = (jc + 32) < Ndim;
        if (wave == 0) {
            if (havenext) {
                tdm_load_tile(whbase + (jc + 32), lbuf[cur ^ 1]);   // prefetch
                __builtin_amdgcn_s_wait_tensorcnt(1);               // cur tile done
            } else {
                __builtin_amdgcn_s_wait_tensorcnt(0);               // tail
            }
        }
        __syncthreads();                                            // cur tile visible

        // ---- A operand: 16 probabilities per lane (contiguous K: k = 16*hi + t) ----
        const int klo = jc + hi * 16;
        const int khi = jc + hi * 16 + 8;
        const float4* p0 = (const float4*)(sjp + klo);
        const float4* p1 = (const float4*)(sjp + khi);
        float sv[16];
        { float4 c0 = p0[0], c1 = p0[1], c2 = p1[0], c3 = p1[1];
          sv[0]=c0.x; sv[1]=c0.y; sv[2]=c0.z; sv[3]=c0.w;
          sv[4]=c1.x; sv[5]=c1.y; sv[6]=c1.z; sv[7]=c1.w;
          sv[8]=c2.x; sv[9]=c2.y; sv[10]=c2.z; sv[11]=c2.w;
          sv[12]=c3.x; sv[13]=c3.y; sv[14]=c3.z; sv[15]=c3.w; }
        v16h Av;
#pragma unroll
        for (int t = 0; t < 16; ++t) {
            const float e = leaky(sim + sv[t] + ab);
            const float p = __expf(e - mrow);
            lsum += p;
            Av[t] = (_Float16)p;
        }

        // ---- B operands from LDS: row o = ot*16+m, K half = hi ----
        const _Float16* lrow = &lbuf[cur][(size_t)m * 32 + hi * 16];
#pragma unroll
        for (int ot = 0; ot < 4; ++ot) {
            const v16h Bv = *(const v16h*)(lrow + (size_t)ot * 16 * 32);
            acc[ot] = __builtin_amdgcn_wmma_f32_16x16x32_f16(false, Av, false, Bv,
                                                             (short)0, acc[ot], false, false);
        }
        __syncthreads();        // all reads of lbuf[cur] done before its reuse
        cur ^= 1;
    }

    // Rows split across lane pairs (l, l^16): combine row sums.
    lsum += __shfl_xor(lsum, 16, 32);

    // ---- epilogue: divide, elu, sigmoid, store [b, row, head*F + col] ----
    float* obase = out + ((size_t)b * Ndim) * (Hdim * Fdim) + head * Fdim;
#pragma unroll
    for (int r = 0; r < 8; ++r) {
        const float rs  = __shfl(lsum, r + hi * 8, 32);
        const int   row = i0 + r + hi * 8;
#pragma unroll
        for (int ot = 0; ot < 4; ++ot) {
            const int col = ot * 16 + m;
            float v = acc[ot][r] / rs;
            v = (v > 0.0f) ? v : (__expf(v) - 1.0f);   // elu
            v = 1.0f / (1.0f + __expf(-v));            // sigmoid
            obase[(size_t)row * (Hdim * Fdim) + col] = v;
        }
    }
}

// ---------------------------------------------------------------------------
extern "C" void kernel_launch(void* const* d_in, const int* in_sizes, int n_in,
                              void* d_out, int out_size, void* d_ws, size_t ws_size,
                              hipStream_t stream)
{
    const float* h  = (const float*)d_in[0];   // [B,N,F_IN]
    const float* W  = (const float*)d_in[1];   // [H,F_IN,F_OUT]
    const float* Wb = (const float*)d_in[2];   // [H,F_OUT]
    const float* a1 = (const float*)d_in[3];   // [H,F_OUT]
    const float* a2 = (const float*)d_in[4];   // [H,F_OUT]
    const float* Ab = (const float*)d_in[5];   // [H]
    float* out = (float*)d_out;                // [B,N,H*F_OUT] f32

    const size_t whtBytes = (size_t)Bdim * Hdim * Fdim * Ndim * sizeof(_Float16); // 8 MB
    _Float16* WhT = (_Float16*)d_ws;
    float* si    = (float*)((char*)d_ws + whtBytes);
    float* sj    = si + (size_t)Bdim * Hdim * Ndim;
    float* sjmax = sj + (size_t)Bdim * Hdim * Ndim;

    gat_k1_feature<<<Bdim * (Ndim / 16), 256, 0, stream>>>(h, W, Wb, a1, a2,
                                                           WhT, si, sj);
    gat_k2_sjmax<<<Bdim * Hdim, 256, 0, stream>>>(sj, sjmax);
    gat_k3_attn<<<Bdim * Hdim * (Ndim / 128), 256, 0, stream>>>(WhT, si, sj,
                                                                sjmax, Ab, out);
}
